// Resampler_5033701670986
// MI455X (gfx1250) — compile-verified, matrix operand order checked by reading, not run on hardware
//
#include <hip/hip_runtime.h>
#include <cstdint>
#include <cstddef>

#define B_    64
#define S_    257
#define ED_   768
#define D_    1024
#define DEPTH_ 8
#define H_    16
#define DH_   64
#define NQ_   32
#define OD_   1024
#define FF_   4096
#define TT_   (S_ + NQ_)   // 289 keys
#define MX_   (B_ * S_)    // 16448 rows (div 64)
#define ML_   (B_ * NQ_)   // 2048 rows  (div 64)

typedef __bf16 bf16;
typedef __attribute__((ext_vector_type(16))) __bf16 bf16x16;
typedef __attribute__((ext_vector_type(8)))  float  floatx8;

// CDNA5 async global->LDS path (bypasses VGPRs, tracked by ASYNCcnt),
// guarded so the build falls back to the proven synchronous copy.
#if defined(__AMDGCN__) && __has_builtin(__builtin_amdgcn_global_load_async_to_lds_b128) && \
    __has_builtin(__builtin_amdgcn_s_wait_asynccnt)
#define USE_ASYNC_LDS 1
#define ASYNC_WAIT() __builtin_amdgcn_s_wait_asynccnt(0)
typedef __attribute__((ext_vector_type(4))) int v4i_;
typedef __attribute__((address_space(1))) v4i_ as1_v4i;
typedef __attribute__((address_space(3))) v4i_ as3_v4i;
#else
#define USE_ASYNC_LDS 0
#define ASYNC_WAIT()
#endif

// ---------------------------------------------------------------------------
// elementwise f32 -> bf16 (vectorized: 4 elems/thread; all sizes are mult. of 4)
// ---------------------------------------------------------------------------
__global__ void cvt_kernel(const float* __restrict__ in, bf16* __restrict__ out, size_t n4) {
    size_t i = (size_t)blockIdx.x * blockDim.x + threadIdx.x;
    if (i < n4) {
        float4 v = ((const float4*)in)[i];
        bf16 o[4] = {(bf16)v.x, (bf16)v.y, (bf16)v.z, (bf16)v.w};
        *(uint2*)(out + 4 * i) = *(uint2*)o;
    }
}

// broadcast latents [NQ*D] -> lat [B*NQ*D], in float4 units (mask4 = NQ*D/4-1)
__global__ void bcast_kernel(const float* __restrict__ src, float* __restrict__ dst,
                             size_t n4, unsigned mask4) {
    size_t i = (size_t)blockIdx.x * blockDim.x + threadIdx.x;
    if (i < n4) ((float4*)dst)[i] = ((const float4*)src)[i & mask4];
}

// ---------------------------------------------------------------------------
// bf16 WMMA GEMM: C[M,N] = A[M,K] * B[K,N]  (both row-major bf16)
// Block tile 64x128, 128 threads (4 waves), wave tile 32x64 (2x4 WMMA tiles)
// Double-buffered LDS; A tile staged via async global->LDS when available.
// EPI: 1 = +bias -> f32 ; 2 = -> bf16 ; 4 = f32 += ; 5 = GELU(exact) -> bf16
// ---------------------------------------------------------------------------
#define TM 64
#define TN 128
#define TK 32
#define LDS_STR 40   // halves; 80 B row stride, 16B-aligned

__device__ __forceinline__ bf16x16 load_frag(const bf16* base, int lane) {
    // lane<16:  row=lane,     elems 0..7 = K 0..7,  elems 8..15 = K 16..23
    // lane>=16: row=lane-16,  elems 0..7 = K 8..15, elems 8..15 = K 24..31
    const bf16* p = base + (lane & 15) * LDS_STR + ((lane >> 4) << 3);
    bf16x16 f;
    ((uint4*)&f)[0] = *(const uint4*)(p);
    ((uint4*)&f)[1] = *(const uint4*)(p + 16);
    return f;
}

// copy 16 bytes global -> LDS (async if the toolchain exposes the builtin)
__device__ __forceinline__ void stage16(const bf16* g, bf16* l) {
#if USE_ASYNC_LDS
    __builtin_amdgcn_global_load_async_to_lds_b128((as1_v4i*)(g), (as3_v4i*)(l), 0, 0);
#else
    *(uint4*)l = *(const uint4*)g;
#endif
}

struct Bregs { union { uint4 v; unsigned short s[8]; } r0a, r0b, r1a, r1b; };

__device__ __forceinline__ void load_bregs(Bregs& br, const bf16* bp, int N) {
    br.r0a.v = *(const uint4*)(bp);
    br.r0b.v = *(const uint4*)(bp + 8);
    br.r1a.v = *(const uint4*)(bp + N);
    br.r1b.v = *(const uint4*)(bp + N + 8);
}

__device__ __forceinline__ void store_bregs(const Bregs& br, bf16* base, int bcol, int brow) {
    #pragma unroll
    for (int j = 0; j < 8; ++j) {
        *(unsigned*)(&base[(bcol + j)     * LDS_STR + brow]) =
            (unsigned)br.r0a.s[j] | ((unsigned)br.r1a.s[j] << 16);
        *(unsigned*)(&base[(bcol + 8 + j) * LDS_STR + brow]) =
            (unsigned)br.r0b.s[j] | ((unsigned)br.r1b.s[j] << 16);
    }
}

template <int EPI>
__launch_bounds__(128)
__global__ void gemm_kernel(const bf16* __restrict__ A, const bf16* __restrict__ Bm,
                            int M, int N, int K,
                            const float* __restrict__ bias,
                            float* __restrict__ Cf, bf16* __restrict__ Cb) {
    __shared__ __align__(16) bf16 As[2][TM * LDS_STR];
    __shared__ __align__(16) bf16 Bt[2][TN * LDS_STR];   // transposed: Bt[n][k]

    const int tid    = threadIdx.x;
    const int blockM = blockIdx.y * TM;
    const int blockN = blockIdx.x * TN;
    const int w      = tid >> 5;
    const int lane   = tid & 31;
    const int wm     = (w >> 1) * 32;
    const int wn     = (w & 1) * 64;

    floatx8 c[2][4] = {};

    // cooperative-load assignments
    const int arow = tid >> 1;            // 0..63
    const int acol = (tid & 1) << 4;      // 0 or 16 (halves)
    const int brow = (tid >> 3) << 1;     // 0,2,...,30
    const int bcol = (tid & 7) << 4;      // 0..112 (halves)
    const int aoff = arow * LDS_STR + acol;

    const bf16* aptr = A  + (size_t)(blockM + arow) * K + acol;
    const bf16* bptr = Bm + (size_t)brow * N + blockN + bcol;

    // prologue: stage tile 0 into buffer 0
    stage16(aptr,     &As[0][aoff]);
    stage16(aptr + 8, &As[0][aoff + 8]);
    {
        Bregs br;
        load_bregs(br, bptr, N);
        store_bregs(br, Bt[0], bcol, brow);
    }
    ASYNC_WAIT();
    __syncthreads();

    const int niter = K / TK;
    for (int it = 0; it < niter; ++it) {
        const int cur = it & 1, nxt = cur ^ 1;
        const bool more = (it + 1) < niter;
        Bregs br;
        if (more) {
            aptr += TK;
            bptr += (size_t)TK * N;
            stage16(aptr,     &As[nxt][aoff]);      // async: overlaps compute below
            stage16(aptr + 8, &As[nxt][aoff + 8]);
            load_bregs(br, bptr, N);
            if (it + 2 < niter) {
                __builtin_prefetch(aptr + TK, 0, 3);
                __builtin_prefetch(bptr + (size_t)TK * N, 0, 3);
            }
        }

        bf16x16 af[2], bfr[4];
        #pragma unroll
        for (int i = 0; i < 2; ++i) af[i]  = load_frag(&As[cur][(wm + 16 * i) * LDS_STR], lane);
        #pragma unroll
        for (int j = 0; j < 4; ++j) bfr[j] = load_frag(&Bt[cur][(wn + 16 * j) * LDS_STR], lane);
        #pragma unroll
        for (int i = 0; i < 2; ++i)
            #pragma unroll
            for (int j = 0; j < 4; ++j)
                c[i][j] = __builtin_amdgcn_wmma_f32_16x16x32_bf16(
                    false, af[i], false, bfr[j], (short)0, c[i][j], false, false);

        if (more) {
            store_bregs(br, Bt[nxt], bcol, brow);   // writes other buffer: no pre-barrier
            ASYNC_WAIT();
        }
        __syncthreads();
    }

    // epilogue: VGPR r -> row r (lanes 0-15) / row 8+r (lanes 16-31); col = lane%16
    const int colw   = lane & 15;
    const int rowoff = (lane >> 4) << 3;
    #pragma unroll
    for (int i = 0; i < 2; ++i) {
        #pragma unroll
        for (int j = 0; j < 4; ++j) {
            const int mb = blockM + wm + 16 * i + rowoff;
            const int nb = blockN + wn + 16 * j + colw;
            #pragma unroll
            for (int r = 0; r < 8; ++r) {
                size_t idx = (size_t)(mb + r) * N + nb;
                float v = c[i][j][r];
                if constexpr (EPI == 1) { Cf[idx] = v + bias[nb]; }
                else if constexpr (EPI == 2) { Cb[idx] = (bf16)v; }
                else if constexpr (EPI == 4) { Cf[idx] += v; }
                else if constexpr (EPI == 5) {
                    float g = 0.5f * v * (1.0f + erff(v * 0.70710678118654752f));
                    Cb[idx] = (bf16)g;
                } else { Cf[idx] = v; }
            }
        }
    }
}

// ---------------------------------------------------------------------------
// LayerNorm over rows of `width` f32; out either bf16 or f32
// ---------------------------------------------------------------------------
template <bool OUT_BF16>
__launch_bounds__(256)
__global__ void ln_kernel(const float* __restrict__ x,
                          const float* __restrict__ sc, const float* __restrict__ bi,
                          int width, float* __restrict__ outf, bf16* __restrict__ outb) {
    const int row = blockIdx.x;
    const float* xr = x + (size_t)row * width;
    float sum = 0.f, sq = 0.f;
    for (int i = threadIdx.x; i < width; i += 256) { float v = xr[i]; sum += v; sq += v * v; }
    #pragma unroll
    for (int o = 16; o; o >>= 1) { sum += __shfl_xor(sum, o, 32); sq += __shfl_xor(sq, o, 32); }
    __shared__ float rs[8], rq[8];
    __shared__ float mu_s, rstd_s;
    const int w = threadIdx.x >> 5;
    if ((threadIdx.x & 31) == 0) { rs[w] = sum; rq[w] = sq; }
    __syncthreads();
    if (threadIdx.x == 0) {
        float ts = 0.f, tq = 0.f;
        #pragma unroll
        for (int i = 0; i < 8; ++i) { ts += rs[i]; tq += rq[i]; }
        float mu  = ts / (float)width;
        float var = tq / (float)width - mu * mu;
        mu_s = mu; rstd_s = rsqrtf(var + 1e-5f);
    }
    __syncthreads();
    const float mu = mu_s, rstd = rstd_s;
    for (int i = threadIdx.x; i < width; i += 256) {
        float v = (xr[i] - mu) * rstd * sc[i] + bi[i];
        if constexpr (OUT_BF16) outb[(size_t)row * width + i] = (bf16)v;
        else                    outf[(size_t)row * width + i] = v;
    }
}

// ---------------------------------------------------------------------------
// Cross-attention: one block per (b,h). q[B,NQ,1024], kv split as
// kvx[B,S,2048] / kvl[B,NQ,2048] (k = cols 0..1023, v = cols 1024..2047).
// o[B,NQ,1024] bf16.
// ---------------------------------------------------------------------------
#define KSTR 66    // padded K row stride in LDS (halves)
#define PSTR 292   // padded prob row stride (halves)

__launch_bounds__(256)
__global__ void attn_kernel(const bf16* __restrict__ qg,
                            const bf16* __restrict__ kvx,
                            const bf16* __restrict__ kvl,
                            bf16* __restrict__ og) {
    const int b = blockIdx.x >> 4;
    const int h = blockIdx.x & 15;
    const int tid = threadIdx.x, w = tid >> 5, lane = tid & 31;

    __shared__ bf16 Ks[TT_ * KSTR];   // 38,148 B
    __shared__ bf16 Qs[NQ_ * DH_];    //  4,096 B
    __shared__ bf16 Ps[NQ_ * PSTR];   // 18,688 B

    for (int idx = tid; idx < TT_ * DH_; idx += 256) {
        int t = idx >> 6, d = idx & 63;
        bf16 kv;
        if (t < S_) kv = kvx[(((size_t)b * S_  + t)        << 11) + (h << 6) + d];
        else        kv = kvl[(((size_t)b * NQ_ + (t - S_)) << 11) + (h << 6) + d];
        Ks[t * KSTR + d] = kv;
    }
    for (int idx = tid; idx < NQ_ * DH_; idx += 256) {
        int qi = idx >> 6, d = idx & 63;
        Qs[idx] = qg[((size_t)b * NQ_ + qi) * 1024 + (h << 6) + d];
    }
    __syncthreads();

    const float scale = 0.125f;   // 1/sqrt(64)
    for (int qi = w; qi < NQ_; qi += 8) {
        float l[10];
        float mx = -3.0e38f;
        int cnt = 0;
        for (int t = lane; t < TT_; t += 32) {
            const bf16* kp = &Ks[t * KSTR];
            const bf16* qp = &Qs[qi * DH_];
            float acc = 0.f;
            #pragma unroll
            for (int d = 0; d < 64; ++d) acc += (float)qp[d] * (float)kp[d];
            acc *= scale;
            l[cnt++] = acc;
            mx = fmaxf(mx, acc);
        }
        #pragma unroll
        for (int o = 16; o; o >>= 1) mx = fmaxf(mx, __shfl_xor(mx, o, 32));
        float sum = 0.f;
        for (int c2 = 0; c2 < cnt; ++c2) { float e = __expf(l[c2] - mx); l[c2] = e; sum += e; }
        #pragma unroll
        for (int o = 16; o; o >>= 1) sum += __shfl_xor(sum, o, 32);
        const float inv = 1.f / sum;
        cnt = 0;
        for (int t = lane; t < TT_; t += 32) Ps[qi * PSTR + t] = (bf16)(l[cnt++] * inv);
    }
    __syncthreads();

    // P @ V: wave w handles rows 4w..4w+3; lane owns dims d0=lane, d1=lane+32
    float acc[4][2] = {};
    const int qbase = w << 2;
    const int d0 = lane, d1 = lane + 32;
    for (int t = 0; t < TT_; ++t) {
        const bf16* vp;
        if (t < S_) vp = kvx + (((size_t)b * S_  + t)        << 11) + 1024 + (h << 6);
        else        vp = kvl + (((size_t)b * NQ_ + (t - S_)) << 11) + 1024 + (h << 6);
        float v0 = (float)vp[d0], v1 = (float)vp[d1];
        #pragma unroll
        for (int r = 0; r < 4; ++r) {
            float p = (float)Ps[(qbase + r) * PSTR + t];
            acc[r][0] += p * v0;
            acc[r][1] += p * v1;
        }
    }
    #pragma unroll
    for (int r = 0; r < 4; ++r) {
        size_t oi = ((size_t)b * NQ_ + qbase + r) * 1024 + (h << 6);
        og[oi + d0] = (bf16)acc[r][0];
        og[oi + d1] = (bf16)acc[r][1];
    }
}

// ---------------------------------------------------------------------------
// host orchestration
// ---------------------------------------------------------------------------
extern "C" void kernel_launch(void* const* d_in, const int* in_sizes, int n_in,
                              void* d_out, int out_size, void* d_ws, size_t ws_size,
                              hipStream_t stream) {
    (void)in_sizes; (void)n_in; (void)out_size; (void)ws_size;
    const float* x    = (const float*)d_in[0];
    const float* lats = (const float*)d_in[1];
    const float* piw  = (const float*)d_in[2];
    const float* pib  = (const float*)d_in[3];
    const float* ln1s = (const float*)d_in[4];
    const float* ln1b = (const float*)d_in[5];
    const float* ln2s = (const float*)d_in[6];
    const float* ln2b = (const float*)d_in[7];
    const float* wq   = (const float*)d_in[8];
    const float* wkv  = (const float*)d_in[9];
    const float* wo   = (const float*)d_in[10];
    const float* ffns = (const float*)d_in[11];
    const float* ffnb = (const float*)d_in[12];
    const float* w1   = (const float*)d_in[13];
    const float* w2   = (const float*)d_in[14];
    const float* pow_ = (const float*)d_in[15];
    const float* pob  = (const float*)d_in[16];
    const float* nos  = (const float*)d_in[17];
    const float* nob  = (const float*)d_in[18];
    float* out = (float*)d_out;

    char* ws = (char*)d_ws;
    size_t off = 0;
    auto alloc = [&](size_t bytes) -> void* {
        off = (off + 255) & ~(size_t)255;
        void* p = ws + off;
        off += bytes;
        return p;
    };
    bf16*  x_bf  = (bf16*) alloc((size_t)MX_ * ED_ * 2);
    float* xp    = (float*)alloc((size_t)MX_ * D_  * 4);
    bf16*  xn_bf = (bf16*) alloc((size_t)MX_ * D_  * 2);
    bf16*  kvx   = (bf16*) alloc((size_t)MX_ * 2048 * 2);
    bf16*  kvl   = (bf16*) alloc((size_t)ML_ * 2048 * 2);
    float* lat   = (float*)alloc((size_t)ML_ * D_  * 4);
    bf16*  lnl   = (bf16*) alloc((size_t)ML_ * D_  * 2);
    bf16*  qb    = (bf16*) alloc((size_t)ML_ * D_  * 2);
    bf16*  ob    = (bf16*) alloc((size_t)ML_ * D_  * 2);
    bf16*  hb    = (bf16*) alloc((size_t)ML_ * D_  * 2);
    bf16*  gbuf  = (bf16*) alloc((size_t)ML_ * FF_ * 2);
    bf16*  latb  = (bf16*) alloc((size_t)ML_ * D_  * 2);
    float* outp  = (float*)alloc((size_t)ML_ * OD_ * 4);
    bf16*  wbuf  = (bf16*) alloc((size_t)D_  * FF_ * 2);   // reusable weight slice

    auto cvt = [&](const float* src, bf16* dst, size_t n) {
        size_t n4 = n / 4;
        cvt_kernel<<<(unsigned)((n4 + 255) / 256), 256, 0, stream>>>(src, dst, n4);
    };
    auto ggrid = [](int M, int N) { return dim3((unsigned)(N / 128), (unsigned)(M / 64)); };

    // proj_in: xp = x @ piw + pib
    cvt(x,   x_bf, (size_t)MX_ * ED_);
    cvt(piw, wbuf, (size_t)ED_ * D_);
    gemm_kernel<1><<<ggrid(MX_, D_), 128, 0, stream>>>(x_bf, wbuf, MX_, D_, ED_, pib, xp, nullptr);

    // lat = broadcast(latents)
    {
        size_t n4 = (size_t)ML_ * D_ / 4;
        bcast_kernel<<<(unsigned)((n4 + 255) / 256), 256, 0, stream>>>(
            lats, lat, n4, (unsigned)(NQ_ * D_ / 4 - 1));
    }

    for (int i = 0; i < DEPTH_; ++i) {
        ln_kernel<true><<<MX_, 256, 0, stream>>>(xp,  ln1s + i * D_, ln1b + i * D_, D_, nullptr, xn_bf);
        ln_kernel<true><<<ML_, 256, 0, stream>>>(lat, ln2s + i * D_, ln2b + i * D_, D_, nullptr, lnl);

        cvt(wq + (size_t)i * D_ * 1024, wbuf, (size_t)D_ * 1024);
        gemm_kernel<2><<<ggrid(ML_, 1024), 128, 0, stream>>>(lnl, wbuf, ML_, 1024, D_, nullptr, nullptr, qb);

        cvt(wkv + (size_t)i * D_ * 2048, wbuf, (size_t)D_ * 2048);
        gemm_kernel<2><<<ggrid(MX_, 2048), 128, 0, stream>>>(xn_bf, wbuf, MX_, 2048, D_, nullptr, nullptr, kvx);
        gemm_kernel<2><<<ggrid(ML_, 2048), 128, 0, stream>>>(lnl,   wbuf, ML_, 2048, D_, nullptr, nullptr, kvl);

        attn_kernel<<<B_ * H_, 256, 0, stream>>>(qb, kvx, kvl, ob);

        cvt(wo + (size_t)i * 1024 * D_, wbuf, (size_t)1024 * D_);
        gemm_kernel<4><<<ggrid(ML_, D_), 128, 0, stream>>>(ob, wbuf, ML_, D_, 1024, nullptr, lat, nullptr);

        ln_kernel<true><<<ML_, 256, 0, stream>>>(lat, ffns + i * D_, ffnb + i * D_, D_, nullptr, hb);

        cvt(w1 + (size_t)i * D_ * FF_, wbuf, (size_t)D_ * FF_);
        gemm_kernel<5><<<ggrid(ML_, FF_), 128, 0, stream>>>(hb, wbuf, ML_, FF_, D_, nullptr, nullptr, gbuf);

        cvt(w2 + (size_t)i * FF_ * D_, wbuf, (size_t)FF_ * D_);
        gemm_kernel<4><<<ggrid(ML_, D_), 128, 0, stream>>>(gbuf, wbuf, ML_, D_, FF_, nullptr, lat, nullptr);
    }

    // proj_out + final LN
    cvt(lat,  latb, (size_t)ML_ * D_);
    cvt(pow_, wbuf, (size_t)D_ * OD_);
    gemm_kernel<1><<<ggrid(ML_, OD_), 128, 0, stream>>>(latb, wbuf, ML_, OD_, D_, pob, outp, nullptr);
    ln_kernel<false><<<ML_, 256, 0, stream>>>(outp, nos, nob, OD_, out, nullptr);
}
